// Net_27573690040374
// MI455X (gfx1250) — matrix operand order repaired and checked by reading, not compiled, not run on hardware
//
#include <hip/hip_runtime.h>
#include <hip/hip_bf16.h>

typedef __attribute__((ext_vector_type(16))) _Float16 v16h;
typedef __attribute__((ext_vector_type(8)))  _Float16 v8h;
typedef __attribute__((ext_vector_type(8)))  float    v8f;

#define TPB 256

// ---------------------------------------------------------------- utilities

__global__ void zero_f32(float* __restrict__ p, int n) {
  int t = blockIdx.x * blockDim.x + threadIdx.x;
  if (t < n) p[t] = 0.0f;
}

__global__ void cvt_f16(const float* __restrict__ src, _Float16* __restrict__ dst, int n) {
  int t = blockIdx.x * blockDim.x + threadIdx.x;
  if (t < n) dst[t] = (_Float16)src[t];
}

// Transpose+convert the four 64x64 weight matrices to f16 [n][k] layout.
__global__ void prep_wt(const float* __restrict__ w0, const float* __restrict__ w1,
                        const float* __restrict__ w2, const float* __restrict__ w3,
                        _Float16* __restrict__ wt) {
  int t = blockIdx.x * blockDim.x + threadIdx.x;   // 4 * 64 * 64
  if (t >= 4 * 4096) return;
  int which = t >> 12, r = t & 4095, n = r >> 6, k = r & 63;
  const float* w = (which == 0) ? w0 : (which == 1) ? w1 : (which == 2) ? w2 : w3;
  wt[t] = (_Float16)w[k * 64 + n];
}

// ---------------------------------------------------------------- degrees

__global__ void deg_g1(const int* __restrict__ dst, float* __restrict__ deg, int E) {
  int e = blockIdx.x * blockDim.x + threadIdx.x;
  if (e < E) atomicAdd(&deg[dst[e]], 1.0f);
}

__global__ void dinv12(const float* __restrict__ deg, float* __restrict__ dinv1,
                       float* __restrict__ dinv2, int n) {
  int i = blockIdx.x * blockDim.x + threadIdx.x;
  if (i >= n) return;
  float d = deg[i];
  dinv1[i] = rsqrtf(d + 1.0f);
  dinv2[i] = rsqrtf(d + 2.0f);
}

__global__ void dinvp_k(const float* __restrict__ deg, float* __restrict__ dinvp, int n) {
  int i = blockIdx.x * blockDim.x + threadIdx.x;
  if (i < n) dinvp[i] = rsqrtf(deg[i] + 2.0f);
}

// ---------------------------------------------------------------- augmented graph (A+I)^2

__global__ void build_dst2(const int* __restrict__ dst, int* __restrict__ dst2,
                           int n, int d, int per) {
  int t = blockIdx.x * blockDim.x + threadIdx.x;
  if (t >= n * per) return;
  int i = t / per, r = t - i * per;
  int a = r / (d + 1), b = r - a * (d + 1);
  int m = (a < d) ? dst[i * d + a] : i;
  int j = (b < d) ? dst[m * d + b] : m;
  dst2[t] = j;
}

__global__ void deg_g2(const int* __restrict__ dst2, const int* __restrict__ sel,
                       float* __restrict__ degp, int n, int per) {
  int t = blockIdx.x * blockDim.x + threadIdx.x;
  if (t >= n * per) return;
  int i = t / per, j = dst2[t];
  if (j != i && sel[i] && sel[j]) atomicAdd(&degp[j], 1.0f);
}

// ---------------------------------------------------------------- dense GEMMs

// Scalar GEMM for the 6->64 input projection.
__global__ void gemm_init(const float* __restrict__ x, const float* __restrict__ W,
                          float* __restrict__ Y, int n) {
  int t = blockIdx.x * blockDim.x + threadIdx.x;
  if (t >= n * 64) return;
  int i = t >> 6, c = t & 63;
  float acc = 0.f;
#pragma unroll
  for (int f = 0; f < 6; ++f) acc += x[i * 6 + f] * W[f * 64 + c];
  Y[t] = acc;
}

// Scalar GEMM for the 64->3 output projection.
__global__ void gemm_out3(const float* __restrict__ h, const float* __restrict__ W,
                          float* __restrict__ Y3, int n) {
  int t = blockIdx.x * blockDim.x + threadIdx.x;
  if (t >= n * 3) return;
  int i = t / 3, c = t - i * 3;
  float acc = 0.f;
#pragma unroll
  for (int k = 0; k < 64; ++k) acc += h[i * 64 + k] * W[k * 3 + c];
  Y3[t] = acc;
}

// WMMA GEMM: Y[N,64] = Ah[N,64] (f16) @ W (Wt is f16, transposed [n][k]).
// One wave -> 16 rows x 64 cols via 8x v_wmma_f32_16x16x32_f16.
__global__ __launch_bounds__(128)
void gemm_h64_wmma(const _Float16* __restrict__ Ah, const _Float16* __restrict__ Wt,
                   float* __restrict__ Y, int nrows) {
  int lane = threadIdx.x & 31;
  int wave = threadIdx.x >> 5;
  int row0 = (blockIdx.x * 4 + wave) * 16;
  if (row0 >= nrows) return;
  int half = lane >> 4;     // lane-half selects K sub-block
  int mr   = lane & 15;     // A: row within tile;  B/C: column within tile

  v8f acc[4] = {};
#pragma unroll
  for (int kt = 0; kt < 2; ++kt) {
    int kbase = kt * 32;
    // A fragment (ISA 16-bit A 16x32): lane(16*half+m) holds
    // K = kbase + half*8 + {0..7}  and  K = kbase + 16 + half*8 + {0..7}
    const _Float16* arow = Ah + (size_t)(row0 + mr) * 64 + kbase + half * 8;
    v8h alo = *(const v8h*)(arow);
    v8h ahi = *(const v8h*)(arow + 16);
    v16h a;
#pragma unroll
    for (int j = 0; j < 8; ++j) { a[j] = alo[j]; a[8 + j] = ahi[j]; }

#pragma unroll
    for (int nt = 0; nt < 4; ++nt) {
      // B fragment (32x16): lane(16*half+n) holds K = kbase + half*8 + {0..7} and K = kbase + 16 + half*8 + {0..7}
      const _Float16* bcol = Wt + (size_t)(nt * 16 + mr) * 64 + kbase + half * 8;
      v8h blo = *(const v8h*)(bcol);
      v8h bhi = *(const v8h*)(bcol + 16);
      v16h b;
#pragma unroll
      for (int j = 0; j < 8; ++j) { b[j] = blo[j]; b[8 + j] = bhi[j]; }

      acc[nt] = __builtin_amdgcn_wmma_f32_16x16x32_f16(
          false, a, false, b, (short)0, acc[nt], false, false);
    }
  }
  // C/D layout: lane(16*half+n), VGPR r -> C[r + 8*half][n]
#pragma unroll
  for (int nt = 0; nt < 4; ++nt)
#pragma unroll
    for (int r = 0; r < 8; ++r)
      Y[(size_t)(row0 + r + 8 * half) * 64 + nt * 16 + mr] = acc[nt][r];
}

// ---------------------------------------------------------------- aggregation

__global__ void agg_g1(const int* __restrict__ dst, const float* __restrict__ Y,
                       const float* __restrict__ dinv, float* __restrict__ out,
                       int E, int D) {
  int t = blockIdx.x * blockDim.x + threadIdx.x;
  if (t >= E * 16) return;
  int e = t >> 4, c4 = (t & 15) << 2;
  int i = e / D, j = dst[e];
  float coef = dinv[i] * dinv[j];
  float4 y = *(const float4*)(Y + (size_t)i * 64 + c4);
  float* o = out + (size_t)j * 64 + c4;
  atomicAdd(o + 0, coef * y.x);
  atomicAdd(o + 1, coef * y.y);
  atomicAdd(o + 2, coef * y.z);
  atomicAdd(o + 3, coef * y.w);
}

__global__ void agg_g2(const int* __restrict__ dst2, const int* __restrict__ sel,
                       const float* __restrict__ Y, const float* __restrict__ dinv,
                       float* __restrict__ out, int n, int per) {
  int t = blockIdx.x * blockDim.x + threadIdx.x;
  if (t >= n * per * 16) return;
  int e = t >> 4, c4 = (t & 15) << 2;
  int i = e / per, j = dst2[e];
  if (j == i || !sel[i] || !sel[j]) return;
  float coef = dinv[i] * dinv[j];
  float4 y = *(const float4*)(Y + (size_t)i * 64 + c4);
  float* o = out + (size_t)j * 64 + c4;
  atomicAdd(o + 0, coef * y.x);
  atomicAdd(o + 1, coef * y.y);
  atomicAdd(o + 2, coef * y.z);
  atomicAdd(o + 3, coef * y.w);
}

__global__ void agg_g1_c3(const int* __restrict__ dst, const float* __restrict__ Y3,
                          const float* __restrict__ dinv, float* __restrict__ out,
                          int E, int D) {
  int e = blockIdx.x * blockDim.x + threadIdx.x;
  if (e >= E) return;
  int i = e / D, j = dst[e];
  float coef = dinv[i] * dinv[j];
  atomicAdd(&out[j * 3 + 0], coef * Y3[i * 3 + 0]);
  atomicAdd(&out[j * 3 + 1], coef * Y3[i * 3 + 1]);
  atomicAdd(&out[j * 3 + 2], coef * Y3[i * 3 + 2]);
}

__global__ void finalize64(float* __restrict__ out, const float* __restrict__ Y,
                           const float* __restrict__ dinv, const float* __restrict__ b,
                           float fill, int do_relu, int n) {
  int t = blockIdx.x * blockDim.x + threadIdx.x;
  if (t >= n * 64) return;
  int i = t >> 6, c = t & 63;
  float dv = dinv[i];
  float v = out[t] + fill * dv * dv * Y[t] + b[c];
  out[t] = do_relu ? fmaxf(v, 0.f) : v;
}

__global__ void finalize3(float* __restrict__ out, const float* __restrict__ Y3,
                          const float* __restrict__ dinv, const float* __restrict__ b, int n) {
  int t = blockIdx.x * blockDim.x + threadIdx.x;
  if (t >= n * 3) return;
  int i = t / 3, c = t - i * 3;
  float dv = dinv[i];
  out[t] = out[t] + dv * dv * Y3[t] + b[c];
}

// ---------------------------------------------------------------- pooling

__global__ void score_k(const float* __restrict__ h0, const float* __restrict__ p,
                        float* __restrict__ s, int n) {
  int i = blockIdx.x * blockDim.x + threadIdx.x;
  if (i >= n) return;
  float num = 0.f, den = 0.f;
#pragma unroll 8
  for (int k = 0; k < 64; ++k) { num += h0[i * 64 + k] * p[k]; den += p[k] * p[k]; }
  s[i] = tanhf(num * rsqrtf(den));
}

// Single-block top-k set selection via threshold bisection (order of perm is
// irrelevant to the final output, only the selected set matters).
__global__ void __launch_bounds__(1024)
topk_select(const float* __restrict__ s, int* __restrict__ sel, int n, int k) {
  __shared__ int cnt;
  __shared__ int base;
  float lo = -1.001f, hi = 1.001f;
  for (int it = 0; it < 50; ++it) {
    float mid = 0.5f * (lo + hi);
    if (threadIdx.x == 0) cnt = 0;
    __syncthreads();
    int c = 0;
    for (int i = threadIdx.x; i < n; i += 1024) c += (s[i] > mid) ? 1 : 0;
    atomicAdd(&cnt, c);
    __syncthreads();
    int total = cnt;
    __syncthreads();
    if (total >= k) lo = mid; else hi = mid;
  }
  if (threadIdx.x == 0) cnt = 0;
  __syncthreads();
  int c = 0;
  for (int i = threadIdx.x; i < n; i += 1024) c += (s[i] > hi) ? 1 : 0;
  atomicAdd(&cnt, c);
  __syncthreads();
  if (threadIdx.x == 0) { base = cnt; cnt = 0; }
  __syncthreads();
  int need = k - base;   // how many boundary (lo,hi] elements to take
  for (int i = threadIdx.x; i < n; i += 1024) {
    float v = s[i];
    int take = (v > hi) ? 1 : 0;
    if (!take && v > lo) {
      int idx = atomicAdd(&cnt, 1);
      take = (idx < need) ? 1 : 0;
    }
    sel[i] = take;
  }
}

__global__ void make_hp(const float* __restrict__ h0, const float* __restrict__ s,
                        const int* __restrict__ sel, float* __restrict__ hp, int n) {
  int t = blockIdx.x * blockDim.x + threadIdx.x;
  if (t >= n * 64) return;
  int i = t >> 6;
  hp[t] = sel[i] ? h0[t] * s[i] : 0.f;
}

__global__ void make_hin2(const float* __restrict__ h0, const float* __restrict__ h1,
                          const int* __restrict__ sel, float* __restrict__ out, int n) {
  int t = blockIdx.x * blockDim.x + threadIdx.x;
  if (t >= n * 64) return;
  int i = t >> 6;
  out[t] = h0[t] + (sel[i] ? h1[t] : 0.f);
}

// ---------------------------------------------------------------- host side

static inline int cdiv(int a, int b) { return (a + b - 1) / b; }

static void run_conv64(const float* X, float* OUT, const _Float16* Wt, const float* bias,
                       const int* dstE, const float* dinv, float fill, int do_relu,
                       float* Y, _Float16* HF, int N, int E, int D, hipStream_t stream) {
  int nb = cdiv(N * 64, TPB);
  cvt_f16<<<nb, TPB, 0, stream>>>(X, HF, N * 64);
  gemm_h64_wmma<<<N / 64, 128, 0, stream>>>(HF, Wt, Y, N);
  zero_f32<<<nb, TPB, 0, stream>>>(OUT, N * 64);
  agg_g1<<<cdiv(E * 16, TPB), TPB, 0, stream>>>(dstE, Y, dinv, OUT, E, D);
  finalize64<<<nb, TPB, 0, stream>>>(OUT, Y, dinv, bias, fill, do_relu, N);
}

extern "C" void kernel_launch(void* const* d_in, const int* in_sizes, int n_in,
                              void* d_out, int out_size, void* d_ws, size_t ws_size,
                              hipStream_t stream) {
  const float* x       = (const float*)d_in[0];
  const int*   eidx    = (const int*)  d_in[1];
  const float* init_W  = (const float*)d_in[3];
  const float* init_b  = (const float*)d_in[4];
  const float* down0_W = (const float*)d_in[5];
  const float* down0_b = (const float*)d_in[6];
  const float* pool_p  = (const float*)d_in[7];
  const float* down1_W = (const float*)d_in[8];
  const float* down1_b = (const float*)d_in[9];
  const float* up0_W   = (const float*)d_in[10];
  const float* up0_b   = (const float*)d_in[11];
  const float* conv1_W = (const float*)d_in[12];
  const float* conv1_b = (const float*)d_in[13];
  const float* conv2_W = (const float*)d_in[14];
  const float* conv2_b = (const float*)d_in[15];
  float* out = (float*)d_out;

  const int N   = in_sizes[0] / 6;      // 16384
  const int E   = in_sizes[1] / 2;      // 131072
  const int D   = E / N;                // 8
  const int per = (D + 1) * (D + 1);    // 81
  const int K   = (N + 1) / 2;          // 8192
  const int* dstE = eidx + E;           // edge_index row 1

  // ---- workspace layout ----
  float*    B1    = (float*)d_ws;             // N*64
  float*    B2    = B1 + (size_t)N * 64;
  float*    B3    = B2 + (size_t)N * 64;
  float*    Y     = B3 + (size_t)N * 64;
  _Float16* HF    = (_Float16*)(Y + (size_t)N * 64);      // N*64 f16
  _Float16* WT    = HF + (size_t)N * 64;                  // 4 * 4096 f16
  int*      DST2  = (int*)(WT + 4 * 4096);                // N*per
  float*    INDEG = (float*)(DST2 + (size_t)N * per);
  float*    DINV1 = INDEG + N;
  float*    DINV2 = DINV1 + N;
  float*    DEGP  = DINV2 + N;
  float*    DINVP = DEGP + N;
  float*    S     = DINVP + N;
  int*      SEL   = (int*)(S + N);

  const _Float16* WT_down0 = WT + 0 * 4096;
  const _Float16* WT_down1 = WT + 1 * 4096;
  const _Float16* WT_up0   = WT + 2 * 4096;
  const _Float16* WT_conv1 = WT + 3 * 4096;

  int nbNH = cdiv(N * 64, TPB);
  int nbN  = cdiv(N, TPB);

  // ---- prep: weights, degrees, augmented graph ----
  prep_wt<<<cdiv(4 * 4096, TPB), TPB, 0, stream>>>(down0_W, down1_W, up0_W, conv1_W, WT);
  zero_f32<<<nbN, TPB, 0, stream>>>(INDEG, N);
  deg_g1<<<cdiv(E, TPB), TPB, 0, stream>>>(dstE, INDEG, E);
  dinv12<<<nbN, TPB, 0, stream>>>(INDEG, DINV1, DINV2, N);
  build_dst2<<<cdiv(N * per, TPB), TPB, 0, stream>>>(dstE, DST2, N, D, per);

  // ---- init conv: h = relu(gcn(x, init_W, fill=1)) -> B1 ----
  gemm_init<<<nbNH, TPB, 0, stream>>>(x, init_W, Y, N);
  zero_f32<<<nbNH, TPB, 0, stream>>>(B1, N * 64);
  agg_g1<<<cdiv(E * 16, TPB), TPB, 0, stream>>>(dstE, Y, DINV1, B1, E, D);
  finalize64<<<nbNH, TPB, 0, stream>>>(B1, Y, DINV1, init_b, 1.0f, 1, N);

  // ---- down0: h0 = relu(gcn(h, fill=2)) -> B2 ----
  run_conv64(B1, B2, WT_down0, down0_b, dstE, DINV2, 2.0f, 1, Y, HF, N, E, D, stream);

  // ---- TopK pooling ----
  score_k<<<nbN, TPB, 0, stream>>>(B2, pool_p, S, N);
  topk_select<<<1, 1024, 0, stream>>>(S, SEL, N, K);
  make_hp<<<nbNH, TPB, 0, stream>>>(B2, S, SEL, B1, N);

  // pooled degrees
  zero_f32<<<nbN, TPB, 0, stream>>>(DEGP, N);
  deg_g2<<<cdiv(N * per, TPB), TPB, 0, stream>>>(DST2, SEL, DEGP, N, per);
  dinvp_k<<<nbN, TPB, 0, stream>>>(DEGP, DINVP, N);

  // ---- down1 (pooled, augmented graph): h1 = relu(gcn(hp, fill=2)) -> B3 ----
  cvt_f16<<<nbNH, TPB, 0, stream>>>(B1, HF, N * 64);
  gemm_h64_wmma<<<N / 64, 128, 0, stream>>>(HF, WT_down1, Y, N);
  zero_f32<<<nbNH, TPB, 0, stream>>>(B3, N * 64);
  agg_g2<<<cdiv(N * per * 16, TPB), TPB, 0, stream>>>(DST2, SEL, Y, DINVP, B3, N, per);
  finalize64<<<nbNH, TPB, 0, stream>>>(B3, Y, DINVP, down1_b, 2.0f, 1, N);

  // ---- unpool + residual: hin2 = h0 + sel*h1 -> B1 ----
  make_hin2<<<nbNH, TPB, 0, stream>>>(B2, B3, SEL, B1, N);

  // ---- up0: h2 = gcn(hin2, fill=2) (no relu) -> B2 ----
  run_conv64(B1, B2, WT_up0, up0_b, dstE, DINV2, 2.0f, 0, Y, HF, N, E, D, stream);

  // ---- conv1: h3 = relu(gcn(h2, fill=1)) -> B1 ----
  run_conv64(B2, B1, WT_conv1, conv1_b, dstE, DINV1, 1.0f, 1, Y, HF, N, E, D, stream);

  // ---- conv2: out = gcn(h3, conv2_W(64->3), fill=1) -> d_out ----
  gemm_out3<<<cdiv(N * 3, TPB), TPB, 0, stream>>>(B1, conv2_W, Y, N);
  zero_f32<<<cdiv(N * 3, TPB), TPB, 0, stream>>>(out, N * 3);
  agg_g1_c3<<<cdiv(E, TPB), TPB, 0, stream>>>(dstE, Y, DINV1, out, E, D);
  finalize3<<<cdiv(N * 3, TPB), TPB, 0, stream>>>(out, Y, DINV1, conv2_b, N);
}
